// SGLUFFCNNet_30812095382066
// MI455X (gfx1250) — compile-verified, matrix operand order checked by reading, not run on hardware
//
#include <hip/hip_runtime.h>
#include <hip/hip_bf16.h>
#include <stdint.h>

// ---------------------------------------------------------------------------
// Problem dims (fixed by reference)
// ---------------------------------------------------------------------------
#define T_STEPS 23
#define BATCH   128
#define HS      512
#define SI      (HS * T_STEPS)   // 11776
#define S2      (SI * 2)         // 23552
#define OUT_N   512
#define K_SPLIT 23               // ffc2 K split: 23552 / 23 = 1024
#define K_SEG   1024

typedef __bf16 v16bf __attribute__((ext_vector_type(16)));
typedef float  v8f   __attribute__((ext_vector_type(8)));

union Frag16 { unsigned int u[8]; v16bf v; };

// packed f32x2 -> bf16x2 via the hardware converter (single VALU op).
// gfx1250 inherits gfx950's V_CVT_PK_BF16_F32; clang declares no builtin for
// it on this target, so emit it directly. dst = {lo16=cvt(a), hi16=cvt(b)}.
__device__ __forceinline__ unsigned int pk_bf16(float a, float b) {
    unsigned int r;
    asm("v_cvt_pk_bf16_f32 %0, %1, %2" : "=v"(r) : "v"(a), "v"(b));
    return r;
}

// K-offset (within a 32-wide K step) of VGPR j for lane-half hl, per the
// documented 16-bit A layout (mirrored for B): groups of 4 VGPRs cover 16 K,
// lanes 0-15 hold the low 8 K of each group, lanes 16-31 the high 8.
__device__ __forceinline__ int frag_k(int j, int hl) {
    return ((j >> 2) << 4) + (hl << 3) + ((j & 3) << 1);
}

// ===========================================================================
// 1) Recurrent scan step: one timestep, 3 fused GEMMs + spiking gating.
//    Transposed activations act[k][b] live in LDS as packed bf16 K-pairs.
//    grid = 8 (batch tiles of 16), block = 256 (8 waves, 4 HS-tiles each).
// ===========================================================================
#define ACT_STRIDE 258   // u32 stride (even for b64 LDS reads, conflict-free)

__device__ __forceinline__ void init_bias(v8f acc[4], const float* __restrict__ bias,
                                          int wave, int hl) {
#pragma unroll
    for (int s = 0; s < 4; ++s)
#pragma unroll
        for (int i = 0; i < 8; ++i)
            acc[s][i] = bias[wave * 64 + s * 16 + i + hl * 8];
}

// acc[s] += W_tile x act_tile over K=512 (bf16 WMMA, f32 accumulate)
__device__ __forceinline__ void gemm_tiles(const unsigned int (*act)[ACT_STRIDE],
                                           const float* __restrict__ W,
                                           v8f acc[4], int wave, int lane) {
    const int nl = lane & 15;
    const int hl = lane >> 4;
#pragma unroll
    for (int s = 0; s < 4; ++s) {
        const int row = wave * 64 + s * 16 + nl;          // HS row of A
        const float* wrow = W + (size_t)row * HS;
        v8f c = acc[s];
        for (int kc = 0; kc < HS; kc += 32) {
            Frag16 a, b;
#pragma unroll
            for (int j = 0; j < 8; j += 2) {
                const int k = kc + frag_k(j, hl);
                float4 f = *(const float4*)(wrow + k);    // contiguous along K
                a.u[j]     = pk_bf16(f.x, f.y);
                a.u[j + 1] = pk_bf16(f.z, f.w);
                b.u[j]     = act[nl][(k >> 1)];
                b.u[j + 1] = act[nl][(k >> 1) + 1];
            }
            c = __builtin_amdgcn_wmma_f32_16x16x32_bf16(false, a.v, false, b.v,
                                                        (short)0, c, false, false);
        }
        acc[s] = c;
    }
}

// spiking gate: pot+=h; fire -> a=pot,pot=0 ; else a=0,pot*=decay
__device__ __forceinline__ void gate(const v8f acc[4], unsigned int (*dst)[ACT_STRIDE],
                                     float* __restrict__ pot,
                                     const float* __restrict__ tresh,
                                     const float* __restrict__ decay,
                                     int wave, int lane, int b0, int first) {
    const int nl = lane & 15, hl = lane >> 4;
    const int b = b0 + nl;
#pragma unroll
    for (int s = 0; s < 4; ++s) {
        float av[8];
#pragma unroll
        for (int i = 0; i < 8; ++i) {
            const int h = wave * 64 + s * 16 + i + hl * 8;
            float p = first ? 0.0f : pot[(size_t)h * BATCH + b];
            p += acc[s][i];
            const bool fire = p > tresh[h];
            av[i] = fire ? p : 0.0f;
            pot[(size_t)h * BATCH + b] = fire ? 0.0f : p * decay[h];
        }
#pragma unroll
        for (int i = 0; i < 8; i += 2) {
            const int h = wave * 64 + s * 16 + i + hl * 8;
            dst[nl][h >> 1] = pk_bf16(av[i], av[i + 1]);
        }
    }
}

__global__ __launch_bounds__(256)
void snn_step_kernel(const float* __restrict__ data,
                     const float* __restrict__ fc1_w, const float* __restrict__ fc1_b,
                     const float* __restrict__ tresh1, const float* __restrict__ decay1,
                     const float* __restrict__ fc2_w, const float* __restrict__ fc2_b,
                     const float* __restrict__ tresh2, const float* __restrict__ decay2,
                     const float* __restrict__ fc3_w, const float* __restrict__ fc3_b,
                     float* __restrict__ pot1, float* __restrict__ pot2,
                     float* __restrict__ gnnT, int t) {
    __shared__ unsigned int actA[16][ACT_STRIDE];
    __shared__ unsigned int actB[16][ACT_STRIDE];

    const int tid  = threadIdx.x;
    const int lane = tid & 31;
    const int wave = tid >> 5;
    const int b0   = blockIdx.x * 16;
    const int nl   = lane & 15;
    const int hl   = lane >> 4;

    // stage x_t^T batch tile -> actA (packed bf16 K-pairs), coalesced along K
    {
        const int bb = tid >> 4;
        const int k0 = (tid & 15) << 5;
        const float* src = data + ((size_t)t * BATCH + (b0 + bb)) * HS + k0;
#pragma unroll
        for (int i = 0; i < 32; i += 4) {
            float4 f = *(const float4*)(src + i);
            actA[bb][((k0 + i) >> 1)]     = pk_bf16(f.x, f.y);
            actA[bb][((k0 + i) >> 1) + 1] = pk_bf16(f.z, f.w);
        }
    }
    __syncthreads();

    v8f acc[4];

    // layer 1
    init_bias(acc, fc1_b, wave, hl);
    gemm_tiles(actA, fc1_w, acc, wave, lane);
    gate(acc, actB, pot1, tresh1, decay1, wave, lane, b0, t == 0);
    __syncthreads();

    // layer 2
    init_bias(acc, fc2_b, wave, hl);
    gemm_tiles(actB, fc2_w, acc, wave, lane);
    gate(acc, actA, pot2, tresh2, decay2, wave, lane, b0, t == 0);
    __syncthreads();

    // layer 3 -> gnnT rows [t*HS .. t*HS+511]
    init_bias(acc, fc3_b, wave, hl);
    gemm_tiles(actA, fc3_w, acc, wave, lane);
#pragma unroll
    for (int s = 0; s < 4; ++s)
#pragma unroll
        for (int i = 0; i < 8; ++i) {
            const int h = wave * 64 + s * 16 + i + hl * 8;
            gnnT[((size_t)t * HS + h) * BATCH + (b0 + nl)] = acc[s][i];
        }
}

// ===========================================================================
// 2) Big transposed GEMM:  out^T[n][b] = W[n][:] . in^T[:][b]  (+bias)(relu)
//    Macro-tile 128n x 128b, K staged through LDS in 64-wide bf16 chunks.
//    blockIdx.y = K-split segment (partials written per segment).
// ===========================================================================
#define WSTRIDE 34   // u32 stride: even (b64 reads) and conflict-free

template <bool RELU, bool BIAS>
__global__ __launch_bounds__(256)
void gemmT_kernel(const float* __restrict__ in,   // [Ktotal][BATCH]
                  const float* __restrict__ W,    // [N][Ktotal]
                  const float* __restrict__ bias, // [N]
                  float* __restrict__ out,        // [ksplits][N][BATCH]
                  int Ktotal, int Ksize) {
    __shared__ unsigned int wbuf[128][WSTRIDE];   // weights, bf16 K-pairs
    __shared__ unsigned int gbuf[128][WSTRIDE];   // activations, bf16 K-pairs

    const int tid  = threadIdx.x;
    const int lane = tid & 31;
    const int wave = tid >> 5;
    const int nl   = lane & 15;
    const int hl   = lane >> 4;
    const int nb0  = blockIdx.x * 128;
    const int kb   = blockIdx.y * Ksize;
    float* outb    = out + (size_t)blockIdx.y * (size_t)gridDim.x * 128 * BATCH;

    const v8f vzero = {0.f, 0.f, 0.f, 0.f, 0.f, 0.f, 0.f, 0.f};
    v8f acc[8];
#pragma unroll
    for (int i = 0; i < 8; ++i) acc[i] = vzero;

    for (int kc = 0; kc < Ksize; kc += 64) {
        // ---- stage weights: 128 rows x 64 K, coalesced 256B per row ----
#pragma unroll
        for (int jj = 0; jj < 8; ++jj) {
            const int idx = jj * 1024 + tid * 4;
            const int row = idx >> 6;
            const int col = idx & 63;
            const float* wp = W + (size_t)(nb0 + row) * Ktotal + kb + kc + col;
            float4 f = *(const float4*)wp;
            wbuf[row][(col >> 1)]     = pk_bf16(f.x, f.y);
            wbuf[row][(col >> 1) + 1] = pk_bf16(f.z, f.w);
            if (kc + 64 < Ksize) __builtin_prefetch(wp + 64, 0, 1);
        }
        // ---- stage activations: pack pairs across K rows ----
#pragma unroll
        for (int jj = 0; jj < 4; ++jj) {
            const int task = jj * 256 + tid;
            const int kp = task >> 5;
            const int bg = (task & 31) << 2;
            const float* s0 = in + (size_t)(kb + kc + 2 * kp) * BATCH + bg;
            float4 fa = *(const float4*)s0;
            float4 fb = *(const float4*)(s0 + BATCH);
            gbuf[bg + 0][kp] = pk_bf16(fa.x, fb.x);
            gbuf[bg + 1][kp] = pk_bf16(fa.y, fb.y);
            gbuf[bg + 2][kp] = pk_bf16(fa.z, fb.z);
            gbuf[bg + 3][kp] = pk_bf16(fa.w, fb.w);
        }
        __syncthreads();

        // ---- compute: 2 K-steps x 8 batch sub-tiles, A fragment reused ----
#pragma unroll
        for (int ks = 0; ks < 2; ++ks) {
            Frag16 a;
            const int row = wave * 16 + nl;
#pragma unroll
            for (int j = 0; j < 8; ++j)
                a.u[j] = wbuf[row][(ks * 32 + frag_k(j, hl)) >> 1];
#pragma unroll
            for (int bs = 0; bs < 8; ++bs) {
                Frag16 b;
#pragma unroll
                for (int j = 0; j < 8; ++j)
                    b.u[j] = gbuf[bs * 16 + nl][(ks * 32 + frag_k(j, hl)) >> 1];
                acc[bs] = __builtin_amdgcn_wmma_f32_16x16x32_bf16(
                    false, a.v, false, b.v, (short)0, acc[bs], false, false);
            }
        }
        __syncthreads();
    }

    // ---- epilogue ----
#pragma unroll
    for (int bs = 0; bs < 8; ++bs)
#pragma unroll
        for (int i = 0; i < 8; ++i) {
            const int n = nb0 + wave * 16 + i + hl * 8;
            float v = acc[bs][i];
            if (BIAS) v += bias[n];
            if (RELU) v = v > 0.f ? v : 0.f;
            outb[(size_t)n * BATCH + bs * 16 + nl] = v;
        }
}

// ===========================================================================
// 3) Deterministic K-split reduction + bias -> final [B][OUT] output
// ===========================================================================
__global__ __launch_bounds__(256)
void reduce_out_kernel(const float* __restrict__ part,  // [K_SPLIT][OUT][BATCH]
                       const float* __restrict__ bias,  // [OUT]
                       float* __restrict__ out) {       // [BATCH][OUT]
    const int e = blockIdx.x * 256 + threadIdx.x;       // over OUT*BATCH
    const int o = e >> 7;
    const int b = e & 127;
    float s = bias[o];
#pragma unroll
    for (int j = 0; j < K_SPLIT; ++j)
        s += part[(size_t)j * OUT_N * BATCH + e];
    out[(size_t)b * OUT_N + o] = s;
}

// ===========================================================================
// launch
// ===========================================================================
extern "C" void kernel_launch(void* const* d_in, const int* in_sizes, int n_in,
                              void* d_out, int out_size, void* d_ws, size_t ws_size,
                              hipStream_t stream) {
    const float* data   = (const float*)d_in[0];
    const float* fc1_w  = (const float*)d_in[1];
    const float* fc1_b  = (const float*)d_in[2];
    const float* tresh1 = (const float*)d_in[3];
    const float* decay1 = (const float*)d_in[4];
    const float* fc2_w  = (const float*)d_in[5];
    const float* fc2_b  = (const float*)d_in[6];
    const float* tresh2 = (const float*)d_in[7];
    const float* decay2 = (const float*)d_in[8];
    const float* fc3_w  = (const float*)d_in[9];
    const float* fc3_b  = (const float*)d_in[10];
    const float* ffc_w  = (const float*)d_in[11];
    const float* ffc_b  = (const float*)d_in[12];
    const float* ffc2_w = (const float*)d_in[13];
    const float* ffc2_b = (const float*)d_in[14];

    // workspace carve-up (~24.6 MB)
    float* pot1 = (float*)d_ws;                    // [HS][BATCH]
    float* pot2 = pot1 + (size_t)HS * BATCH;       // [HS][BATCH]
    float* gnnT = pot2 + (size_t)HS * BATCH;       // [SI][BATCH]
    float* f1T  = gnnT + (size_t)SI * BATCH;       // [S2][BATCH]
    float* part = f1T + (size_t)S2 * BATCH;        // [K_SPLIT][OUT][BATCH]

    // recurrent scan: 23 dependent steps (t==0 treats potentials as zero)
    for (int t = 0; t < T_STEPS; ++t)
        snn_step_kernel<<<dim3(8), dim3(256), 0, stream>>>(
            data, fc1_w, fc1_b, tresh1, decay1, fc2_w, fc2_b, tresh2, decay2,
            fc3_w, fc3_b, pot1, pot2, gnnT, t);

    // f1^T = relu(ffc_w . gnn^T + ffc_b) : weights (1.1 GB) streamed once
    gemmT_kernel<true, true><<<dim3(S2 / 128, 1), dim3(256), 0, stream>>>(
        gnnT, ffc_w, ffc_b, f1T, SI, SI);

    // out^T partials = ffc2_w . f1^T  (23-way deterministic K split)
    gemmT_kernel<false, false><<<dim3(OUT_N / 128, K_SPLIT), dim3(256), 0, stream>>>(
        f1T, ffc2_w, ffc2_b, part, S2, K_SEG);

    // sum partials + bias, de-transpose to [B][OUT]
    reduce_out_kernel<<<dim3((OUT_N * BATCH) / 256), dim3(256), 0, stream>>>(
        part, ffc2_b, (float*)d_out);
}
